// GraphWrapper_54992761258286
// MI455X (gfx1250) — matrix operand order repaired and checked by reading, not compiled, not run on hardware
//
#include <hip/hip_runtime.h>
#include <hip/hip_bf16.h>

typedef __bf16 bf16_t;
typedef __attribute__((ext_vector_type(16))) __bf16 v16bf;
typedef __attribute__((ext_vector_type(8)))  __bf16 v8bf;
typedef __attribute__((ext_vector_type(8)))  float  v8f;

#define HID     256
#define NGRAPH  256
#define NBIO    512
#define NLAYERS 3
#define EPSV    1e-5f

// ---------------------------------------------------------------------------
// WMMA GEMM:  C[M,256] = act(A[M,K](bf16) @ Wt^T + bias)
// Wt is pre-transposed: Wt[n*K + k] = W[k][n]  (bf16), so B fragments are
// contiguous. Block = 256 thr (8 waves); block tile 64x256; wave tile 32x64.
// Epilogue goes through LDS so global stores are coalesced b128.
// NOTE: unconditional prefetch runs at most 128B past a row's end; all A
// buffers are interior regions of the single workspace allocation, so the
// prefetched address is always mapped.
// ---------------------------------------------------------------------------
template <bool RELU, bool OUT_BF16>
__global__ __launch_bounds__(256) void gemm_n256(
    const bf16_t* __restrict__ A, const bf16_t* __restrict__ Wt,
    const float* __restrict__ bias, void* __restrict__ out, int M, int K) {
  constexpr int OB = OUT_BF16 ? 2 : 4;          // output bytes/elem
  __shared__ __align__(16) char smem[64 * 256 * OB];

  const int tid  = threadIdx.x;
  const int lane = tid & 31;
  const int wave = tid >> 5;   // 0..7
  const int wm   = wave >> 2;  // 0..1  (rows)
  const int wn   = wave & 3;   // 0..3  (cols)
  const int l16  = lane & 15;
  const int lhi  = lane >> 4;  // 0/1

  const int row0 = blockIdx.x * 64 + wm * 32;
  const int col0 = wn * 64;

  v8f acc[2][4];
#pragma unroll
  for (int i = 0; i < 2; ++i)
#pragma unroll
    for (int j = 0; j < 4; ++j) acc[i][j] = (v8f){0, 0, 0, 0, 0, 0, 0, 0};

  // Hoisted, clamped A row pointers (rows >= M are never stored).
  const bf16_t* ap[2];
#pragma unroll
  for (int mf = 0; mf < 2; ++mf) {
    int r = row0 + mf * 16 + l16;
    if (r >= M) r = M - 1;
    ap[mf] = A + (size_t)r * K + lhi * 16;
  }
  const bf16_t* bp[4];
#pragma unroll
  for (int nf = 0; nf < 4; ++nf)
    bp[nf] = Wt + (size_t)(col0 + nf * 16 + l16) * K + lhi * 16;

  for (int k0 = 0; k0 < K; k0 += 32) {
    // A fragments: 16x32 bf16, contiguous K order matching B.
    // lanes 0-15 hold K k0..k0+15 (elems 0-15);
    // lanes 16-31 hold K k0+16..31  -> base offset lhi*16.
    v16bf a[2];
#pragma unroll
    for (int mf = 0; mf < 2; ++mf) {
      v8bf lo = *(const v8bf*)(ap[mf] + k0);
      v8bf hi = *(const v8bf*)(ap[mf] + k0 + 8);
      a[mf] = __builtin_shufflevector(lo, hi, 0, 1, 2, 3, 4, 5, 6, 7, 8, 9, 10,
                                      11, 12, 13, 14, 15);
      __builtin_prefetch(ap[mf] + k0 + 64, 0, 1);  // K-tile after next (always mapped)
    }
    // B fragments: 32x16; lane = column, 16 consecutive K values per lane-half
    // (lanes 0-15: K k0..k0+15, lanes 16-31: K k0+16..31) -> contiguous in Wt.
    v16bf b[4];
#pragma unroll
    for (int nf = 0; nf < 4; ++nf) b[nf] = *(const v16bf*)(bp[nf] + k0);
#pragma unroll
    for (int mf = 0; mf < 2; ++mf)
#pragma unroll
      for (int nf = 0; nf < 4; ++nf)
        acc[mf][nf] = __builtin_amdgcn_wmma_f32_16x16x32_bf16(
            false, a[mf], false, b[nf], (short)0, acc[mf][nf], false, false);
  }

  // ---- epilogue: bias + act into LDS tile (row-major [64][256]) -----------
#pragma unroll
  for (int mf = 0; mf < 2; ++mf) {
#pragma unroll
    for (int nf = 0; nf < 4; ++nf) {
      int c = col0 + nf * 16 + l16;
      float bv = bias[c];
#pragma unroll
      for (int i = 0; i < 8; ++i) {
        int rl = wm * 32 + mf * 16 + lhi * 8 + i;  // local row 0..63
        float v = acc[mf][nf][i] + bv;
        if (RELU) v = v > 0.f ? v : 0.f;
        if (OUT_BF16)
          ((bf16_t*)smem)[rl * 256 + c] = (bf16_t)v;
        else
          ((float*)smem)[rl * 256 + c] = v;
      }
    }
  }
  __syncthreads();

  // ---- coalesced b128 stores of the tile ----------------------------------
  constexpr int CH   = (256 * OB) / 16;  // 16B chunks per row (32 or 64)
  constexpr int ITER = (64 * CH) / 256;  // chunks per thread (8 or 16)
#pragma unroll
  for (int it = 0; it < ITER; ++it) {
    int chunk = it * 256 + tid;
    int r = chunk / CH, cc = chunk % CH;
    int gr = blockIdx.x * 64 + r;
    if (gr < M)
      *(uint4*)((char*)out + (size_t)gr * 256 * OB + cc * 16) =
          *(const uint4*)(smem + (size_t)chunk * 16);
  }
}

// ---------------------------------------------------------------------------
// Elementwise / gather / scatter / segment kernels
// ---------------------------------------------------------------------------
__global__ __launch_bounds__(256) void k_zero_f32(float* p, size_t n) {
  size_t i = (size_t)blockIdx.x * 256 + threadIdx.x;
  if (i < n) p[i] = 0.f;
}

__global__ __launch_bounds__(256) void k_node_embed(
    const int* __restrict__ x, const float* __restrict__ W,
    float* __restrict__ h, size_t nElem) {
  size_t i = (size_t)blockIdx.x * 256 + threadIdx.x;
  if (i >= nElem) return;
  size_t n = i >> 8;
  int c = (int)(i & 255);
  h[i] = W[(size_t)x[n] * HID + c];
}

__global__ __launch_bounds__(256) void k_edge_embed(
    const float* __restrict__ attr, const float* __restrict__ W,
    const float* __restrict__ b, const float* __restrict__ sscale,
    bf16_t* __restrict__ eb, float* __restrict__ scale_row, int E) {
  int i = blockIdx.x, c = threadIdx.x;
  if (i >= E) return;
  float a0 = attr[i * 4 + 0], a1 = attr[i * 4 + 1];
  float a2 = attr[i * 4 + 2], a3 = attr[i * 4 + 3];
  float v = a0 * W[c] + a1 * W[HID + c] + a2 * W[2 * HID + c] +
            a3 * W[3 * HID + c] + b[c];
  eb[(size_t)i * HID + c] = (bf16_t)v;
  if (c == 0) scale_row[i] = (a1 > 0.f) ? sscale[0] : 1.f;
}

__global__ __launch_bounds__(256) void k_transpose_w(
    const float* __restrict__ W, bf16_t* __restrict__ Wt, int K) {
  // W[K,256] row-major -> Wt[256,K] bf16
  size_t i = (size_t)blockIdx.x * 256 + threadIdx.x;
  if (i >= (size_t)K * HID) return;
  int k = (int)(i >> 8), n = (int)(i & 255);
  Wt[(size_t)n * K + k] = (bf16_t)W[i];
}

__global__ __launch_bounds__(256) void k_scale_rows(bf16_t* e, const float* sr,
                                                    size_t n) {
  size_t i = (size_t)blockIdx.x * 256 + threadIdx.x;
  if (i >= n) return;
  e[i] = (bf16_t)((float)e[i] * sr[i >> 8]);
}

__global__ __launch_bounds__(256) void k_message(
    const float* __restrict__ h, const bf16_t* __restrict__ e,
    const int* __restrict__ ei, float* __restrict__ agg, int E) {
  int i = blockIdx.x, c = threadIdx.x;
  if (i >= E) return;
  int s = ei[i], d = ei[E + i];
  float m = h[(size_t)s * HID + c] + (float)e[(size_t)i * HID + c];
  m = m > 0.f ? m : 0.f;
  __hip_atomic_fetch_add(&agg[(size_t)d * HID + c], m, __ATOMIC_RELAXED,
                         __HIP_MEMORY_SCOPE_AGENT);
}

__global__ __launch_bounds__(256) void k_build_z(const float* __restrict__ h,
                                                 const float* __restrict__ agg,
                                                 const float* __restrict__ epsp,
                                                 bf16_t* __restrict__ z,
                                                 size_t n) {
  size_t i = (size_t)blockIdx.x * 256 + threadIdx.x;
  if (i >= n) return;
  float ep = 1.f + epsp[0];
  z[i] = (bf16_t)(ep * h[i] + agg[i]);
}

__global__ void k_bounds(const int* __restrict__ batch, int N, int* gs) {
  int g = threadIdx.x;
  if (g > NGRAPH) return;
  int lo = 0, hi = N;  // first index with batch[i] >= g (batch is sorted)
  while (lo < hi) {
    int mid = (lo + hi) >> 1;
    if (batch[mid] < g) lo = mid + 1; else hi = mid;
  }
  gs[g] = lo;
}

__global__ __launch_bounds__(256) void k_seg_mean(const float* __restrict__ src,
                                                  const int* __restrict__ gs,
                                                  float* __restrict__ dst) {
  int g = blockIdx.x, c = threadIdx.x;
  int s = gs[g], t = gs[g + 1];
  float sum = 0.f;
  for (int n = s; n < t; ++n) sum += src[(size_t)n * HID + c];
  dst[(size_t)g * HID + c] = sum / fmaxf((float)(t - s), 1.f);
}

__global__ __launch_bounds__(256) void k_seg_var(
    const float* __restrict__ h, const int* __restrict__ gs,
    const float* __restrict__ mean, const float* __restrict__ alpha,
    float* __restrict__ var) {
  int g = blockIdx.x, c = threadIdx.x;
  int s = gs[g], t = gs[g + 1];
  float mu = mean[(size_t)g * HID + c] * alpha[c];
  float sum = 0.f;
  for (int n = s; n < t; ++n) {
    float d = h[(size_t)n * HID + c] - mu;
    sum += d * d;
  }
  var[(size_t)g * HID + c] = sum / fmaxf((float)(t - s), 1.f);
}

__global__ __launch_bounds__(256) void k_norm_apply(
    float* __restrict__ h, const int* __restrict__ batch,
    const float* __restrict__ mean, const float* __restrict__ var,
    const float* __restrict__ alpha, const float* __restrict__ gamma,
    const float* __restrict__ beta, size_t n) {
  size_t i = (size_t)blockIdx.x * 256 + threadIdx.x;
  if (i >= n) return;
  int c = (int)(i & 255);
  int g = batch[i >> 8];
  float hc = h[i] - alpha[c] * mean[(size_t)g * HID + c];
  h[i] = gamma[c] * hc * rsqrtf(var[(size_t)g * HID + c] + EPSV) + beta[c];
}

__global__ __launch_bounds__(256) void k_combined(const float* __restrict__ g,
                                                  const float* __restrict__ bio,
                                                  bf16_t* __restrict__ comb) {
  size_t i = (size_t)blockIdx.x * 256 + threadIdx.x;
  const int KH = HID + NBIO;
  if (i >= (size_t)NGRAPH * KH) return;
  int gi = (int)(i / KH), c = (int)(i % KH);
  float v = (c < HID) ? g[(size_t)gi * HID + c] : bio[c - HID];
  comb[i] = (bf16_t)v;
}

__global__ __launch_bounds__(256) void k_head2(const bf16_t* __restrict__ tg,
                                               const float* __restrict__ W2,
                                               const float* __restrict__ b2,
                                               float* __restrict__ out) {
  __shared__ float red[256];
  int g = blockIdx.x, c = threadIdx.x;
  red[c] = (float)tg[(size_t)g * HID + c] * W2[c];
  __syncthreads();
  for (int s = 128; s > 0; s >>= 1) {
    if (c < s) red[c] += red[c + s];
    __syncthreads();
  }
  if (c == 0) out[g] = red[0] + b2[0];
}

// ---------------------------------------------------------------------------
extern "C" void kernel_launch(void* const* d_in, const int* in_sizes, int n_in,
                              void* d_out, int out_size, void* d_ws,
                              size_t ws_size, hipStream_t stream) {
  const int*   x            = (const int*)d_in[0];
  const int*   edge_index   = (const int*)d_in[1];
  const float* edge_attr    = (const float*)d_in[2];
  const int*   batch        = (const int*)d_in[3];
  const float* node_emb_W   = (const float*)d_in[4];
  const float* edge_emb_W   = (const float*)d_in[5];
  const float* edge_emb_b   = (const float*)d_in[6];
  const float* edge_mlp_W1  = (const float*)d_in[7];
  const float* edge_mlp_b1  = (const float*)d_in[8];
  const float* edge_mlp_W2  = (const float*)d_in[9];
  const float* edge_mlp_b2  = (const float*)d_in[10];
  const float* struct_scale = (const float*)d_in[11];
  const float* conv_W1      = (const float*)d_in[12];
  const float* conv_b1      = (const float*)d_in[13];
  const float* conv_W2      = (const float*)d_in[14];
  const float* conv_b2      = (const float*)d_in[15];
  const float* conv_eps     = (const float*)d_in[16];
  const float* norm_gamma   = (const float*)d_in[17];
  const float* norm_beta    = (const float*)d_in[18];
  const float* norm_alpha   = (const float*)d_in[19];
  const float* mean_bio     = (const float*)d_in[20];
  const float* head_W1      = (const float*)d_in[21];
  const float* head_b1      = (const float*)d_in[22];
  const float* head_W2      = (const float*)d_in[23];
  const float* head_b2      = (const float*)d_in[24];
  float* out = (float*)d_out;
  (void)n_in; (void)out_size; (void)ws_size;

  const size_t E  = (size_t)in_sizes[1] / 2;
  const size_t NN = (size_t)in_sizes[0];
  const size_t H  = HID;
  const int    KH = HID + NBIO;  // 768

  auto al = [](size_t v) { return (v + 255) & ~(size_t)255; };
  char* base = (char*)d_ws;
  size_t szE  = al(E * H * sizeof(bf16_t));
  size_t need = al(NN * H * 4) + 2 * al(NN * H * 2);
  size_t szD  = szE > need ? szE : need;

  size_t off = 0;
  bf16_t* eBuf = (bf16_t*)(base + off); off += szE;   // eb, then scaled e
  char*   Dreg = base + off;            off += szD;   // t_e, then agg/z/tN
  bf16_t* tE   = (bf16_t*)Dreg;
  float*  agg  = (float*)Dreg;
  bf16_t* zB   = (bf16_t*)(Dreg + al(NN * H * 4));
  bf16_t* tN   = (bf16_t*)(Dreg + al(NN * H * 4) + al(NN * H * 2));
  float*  hN   = (float*)(base + off);  off += al(NN * H * 4);
  float*  scale_row = (float*)(base + off); off += al(E * 4);
  bf16_t* WtE1 = (bf16_t*)(base + off); off += al(H * H * 2);
  bf16_t* WtE2 = (bf16_t*)(base + off); off += al(H * H * 2);
  bf16_t* WtC1 = (bf16_t*)(base + off); off += al(3 * H * H * 2);
  bf16_t* WtC2 = (bf16_t*)(base + off); off += al(3 * H * H * 2);
  bf16_t* WtH  = (bf16_t*)(base + off); off += al((size_t)KH * H * 2);
  float*  meanB = (float*)(base + off); off += al((size_t)NGRAPH * H * 4);
  float*  varB  = (float*)(base + off); off += al((size_t)NGRAPH * H * 4);
  float*  gpool = (float*)(base + off); off += al((size_t)NGRAPH * H * 4);
  bf16_t* comb  = (bf16_t*)(base + off); off += al((size_t)NGRAPH * KH * 2);
  bf16_t* tG    = (bf16_t*)(base + off); off += al((size_t)NGRAPH * H * 2);
  int*    gs    = (int*)(base + off);    off += al((NGRAPH + 1) * 4);

  dim3 b256(256);
  unsigned gHH = (unsigned)((H * H + 255) / 256);

  // Weight prep (bf16 transposed copies)
  k_transpose_w<<<gHH, b256, 0, stream>>>(edge_mlp_W1, WtE1, (int)H);
  k_transpose_w<<<gHH, b256, 0, stream>>>(edge_mlp_W2, WtE2, (int)H);
  for (int l = 0; l < NLAYERS; ++l) {
    k_transpose_w<<<gHH, b256, 0, stream>>>(conv_W1 + (size_t)l * H * H,
                                            WtC1 + (size_t)l * H * H, (int)H);
    k_transpose_w<<<gHH, b256, 0, stream>>>(conv_W2 + (size_t)l * H * H,
                                            WtC2 + (size_t)l * H * H, (int)H);
  }
  k_transpose_w<<<(unsigned)(((size_t)KH * H + 255) / 256), b256, 0, stream>>>(
      head_W1, WtH, KH);

  // Node embedding + edge embedding
  k_node_embed<<<(unsigned)((NN * H + 255) / 256), b256, 0, stream>>>(
      x, node_emb_W, hN, NN * H);
  k_edge_embed<<<(unsigned)E, b256, 0, stream>>>(
      edge_attr, edge_emb_W, edge_emb_b, struct_scale, eBuf, scale_row, (int)E);

  // Edge MLP: two WMMA GEMMs, then row scaling
  unsigned gE = (unsigned)((E + 63) / 64);
  unsigned gN = (unsigned)((NN + 63) / 64);
  gemm_n256<true, true><<<gE, b256, 0, stream>>>(eBuf, WtE1, edge_mlp_b1, tE,
                                                 (int)E, HID);
  gemm_n256<false, true><<<gE, b256, 0, stream>>>(tE, WtE2, edge_mlp_b2, eBuf,
                                                  (int)E, HID);
  k_scale_rows<<<(unsigned)((E * H + 255) / 256), b256, 0, stream>>>(
      eBuf, scale_row, E * H);

  // Per-graph node ranges (batch is sorted)
  k_bounds<<<1, 288, 0, stream>>>(batch, (int)NN, gs);

  // GINE conv layers
  for (int l = 0; l < NLAYERS; ++l) {
    k_zero_f32<<<(unsigned)((NN * H + 255) / 256), b256, 0, stream>>>(agg,
                                                                      NN * H);
    k_message<<<(unsigned)E, b256, 0, stream>>>(hN, eBuf, edge_index, agg,
                                                (int)E);
    k_build_z<<<(unsigned)((NN * H + 255) / 256), b256, 0, stream>>>(
        hN, agg, conv_eps + l, zB, NN * H);
    gemm_n256<true, true><<<gN, b256, 0, stream>>>(
        zB, WtC1 + (size_t)l * H * H, conv_b1 + (size_t)l * H, tN, (int)NN,
        HID);
    gemm_n256<false, false><<<gN, b256, 0, stream>>>(
        tN, WtC2 + (size_t)l * H * H, conv_b2 + (size_t)l * H, hN, (int)NN,
        HID);
    k_seg_mean<<<NGRAPH, b256, 0, stream>>>(hN, gs, meanB);
    k_seg_var<<<NGRAPH, b256, 0, stream>>>(hN, gs, meanB,
                                           norm_alpha + (size_t)l * H, varB);
    k_norm_apply<<<(unsigned)((NN * H + 255) / 256), b256, 0, stream>>>(
        hN, batch, meanB, varB, norm_alpha + (size_t)l * H,
        norm_gamma + (size_t)l * H, norm_beta + (size_t)l * H, NN * H);
  }

  // Pool + head
  k_seg_mean<<<NGRAPH, b256, 0, stream>>>(hN, gs, gpool);
  k_combined<<<(unsigned)(((size_t)NGRAPH * KH + 255) / 256), b256, 0,
               stream>>>(gpool, mean_bio, comb);
  gemm_n256<true, true><<<(NGRAPH + 63) / 64, b256, 0, stream>>>(
      comb, WtH, head_b1, tG, NGRAPH, KH);
  k_head2<<<NGRAPH, b256, 0, stream>>>(tG, head_W2, head_b2, out);
}
